// SparseBertSelfAttention_21311627722886
// MI455X (gfx1250) — matrix operand order repaired and checked by reading, not compiled, not run on hardware
//
#include <hip/hip_runtime.h>

// ---------------------------------------------------------------------------
// Types for CDNA5 WMMA (gfx1250, wave32)
// ---------------------------------------------------------------------------
typedef __bf16          v16bf __attribute__((ext_vector_type(16)));
typedef float           v8f   __attribute__((ext_vector_type(8)));
typedef unsigned short  v8u16 __attribute__((ext_vector_type(8)));

union BFrag {            // 16 bf16 values per lane = one WMMA A or B operand
    v16bf f;
    v8u16 h[2];          // two 16-byte chunks -> global_load_b128 / ds_load_b128
};

static __device__ __forceinline__ unsigned short f2bf(float x) {
    unsigned int u = __float_as_uint(x);
    unsigned int r = u + 0x7FFFu + ((u >> 16) & 1u);   // round-to-nearest-even
    return (unsigned short)(r >> 16);
}

static __device__ __forceinline__ v8f wmma_bf16(const BFrag& a, const BFrag& b, v8f c) {
    // D = A(16x32 bf16) * B(32x16 bf16) + C(16x16 f32)
    return __builtin_amdgcn_wmma_f32_16x16x32_bf16(
        /*neg_a=*/false, a.f, /*neg_b=*/false, b.f,
        /*c_mod=*/(short)0, c, /*reuse_a=*/false, /*reuse_b=*/false);
}

// ---------------------------------------------------------------------------
// Problem constants
// ---------------------------------------------------------------------------
#define NB     8
#define LQ     1024
#define NH     12
#define DH     64
#define HID    768           // NH * DH
#define NTOK   (NB * LQ)     // 8192

// ---------------------------------------------------------------------------
// Kernel 0: fp32 -> bf16 conversion (vectorized, 8 elems / thread)
// ---------------------------------------------------------------------------
__global__ __launch_bounds__(256) void cvt_f32_bf16(const float* __restrict__ in,
                                                    unsigned short* __restrict__ out,
                                                    int n) {
    int i = (blockIdx.x * 256 + threadIdx.x) * 8;
    if (i + 8 <= n) {
        float4 a = *(const float4*)(in + i);
        float4 b = *(const float4*)(in + i + 4);
        v8u16 o;
        o[0] = f2bf(a.x); o[1] = f2bf(a.y); o[2] = f2bf(a.z); o[3] = f2bf(a.w);
        o[4] = f2bf(b.x); o[5] = f2bf(b.y); o[6] = f2bf(b.z); o[7] = f2bf(b.w);
        *(v8u16*)(out + i) = o;
    }
}

// ---------------------------------------------------------------------------
// Kernel 1: QKV projection  y = x @ W^T + b   (M=8192, N=768, K=768, bf16 WMMA)
//   blockIdx.y selects Q / K / V.  Each wave owns a 16x64 output tile.
//   Q,K stored bf16 [B,H,L,D]; V stored bf16 transposed [B,H,D,L].
// ---------------------------------------------------------------------------
__global__ __launch_bounds__(256) void qkv_gemm(
        const unsigned short* __restrict__ xb,
        const unsigned short* __restrict__ wqb,
        const unsigned short* __restrict__ wkb,
        const unsigned short* __restrict__ wvb,
        const float* __restrict__ bq,
        const float* __restrict__ bk,
        const float* __restrict__ bv,
        unsigned short* __restrict__ qo,
        unsigned short* __restrict__ ko,
        unsigned short* __restrict__ vto) {

    const int mode = blockIdx.y;                 // 0=Q 1=K 2=V
    const unsigned short* W    = (mode == 0) ? wqb : (mode == 1) ? wkb : wvb;
    const float*          bias = (mode == 0) ? bq  : (mode == 1) ? bk  : bv;

    const int tid  = threadIdx.x;
    const int wv_  = tid >> 5;                   // wave id 0..7
    const int lane = tid & 31;
    const int lm   = lane & 15;                  // M (A) / N (B,C/D) index
    const int lh   = lane >> 4;                  // lane half

    const int id = blockIdx.x * 8 + wv_;         // 6144 wave-tiles
    const int tm = id / 12;                      // M tile (16 rows)
    const int tn = id - tm * 12;                 // N tile (64 cols) == head id
    const int m0 = tm * 16;

    v8f acc[4] = {};

    // A-fragment row pointer for this lane (16-bit A layout, ISA 7.12.2):
    // lane-half lh holds K in [lh*8, lh*8+8) and [16+lh*8, 16+lh*8+8)
    const unsigned short* aptr = xb + (size_t)(m0 + lm) * HID + lh * 8;

    for (int k0 = 0; k0 < HID; k0 += 32) {
        BFrag A;
        A.h[0] = *(const v8u16*)(aptr + k0);
        A.h[1] = *(const v8u16*)(aptr + k0 + 16);
#pragma unroll
        for (int j = 0; j < 4; ++j) {
            // B layout: lane = N column, K in [lh*8, lh*8+8) and [16+lh*8, 16+lh*8+8)
            const unsigned short* bp =
                W + (size_t)(tn * 64 + j * 16 + lm) * HID + k0 + lh * 8;
            BFrag Bf;
            Bf.h[0] = *(const v8u16*)bp;
            Bf.h[1] = *(const v8u16*)(bp + 16);
            acc[j] = wmma_bf16(A, Bf, acc[j]);
        }
    }

    // C/D layout: lane lm = N, rows M = lh*8 + r
    const int mb = lh * 8;
    if (mode < 2) {
        // Q or K: [B,H,L,D]
#pragma unroll
        for (int j = 0; j < 4; ++j) {
            const float bb = bias[tn * 64 + j * 16 + lm];
            unsigned short* outp = (mode == 0) ? qo : ko;
#pragma unroll
            for (int r = 0; r < 8; ++r) {
                int m  = m0 + mb + r;
                int bI = m >> 10;
                int li = m & (LQ - 1);
                outp[((size_t)(bI * NH + tn) * LQ + li) * DH + j * 16 + lm] =
                    f2bf(acc[j][r] + bb);
            }
        }
    } else {
        // V transposed: [B,H,D,L] -> rows contiguous, 16B vector store
        int m  = m0 + mb;
        int bI = m >> 10;
        int li = m & (LQ - 1);
#pragma unroll
        for (int j = 0; j < 4; ++j) {
            const float bb = bias[tn * 64 + j * 16 + lm];
            v8u16 pk;
#pragma unroll
            for (int r = 0; r < 8; ++r) pk[r] = f2bf(acc[j][r] + bb);
            *(v8u16*)(vto + ((size_t)(bI * NH + tn) * DH + j * 16 + lm) * LQ + li) = pk;
        }
    }
}

// ---------------------------------------------------------------------------
// Kernel 2: fused attention.  One block = 16 query rows of one (b,h).
//   scores (WMMA) -> LDS fp32 -> softmax / bisection-sparsemax -> LDS bf16 ->
//   context (WMMA, K split across waves) -> LDS reduction -> out fp32
// ---------------------------------------------------------------------------
__global__ __launch_bounds__(256) void attn_fused(
        const unsigned short* __restrict__ qb,
        const unsigned short* __restrict__ kb,
        const unsigned short* __restrict__ vtb,
        const unsigned char*  __restrict__ sparse_mask,
        float* __restrict__ out) {

    __shared__ float          S[16 * LQ];        // 64 KB: fp32 scores / probs; reused for partials
    __shared__ unsigned short Sh[16 * LQ];       // 32 KB: bf16 probs

    const int tid  = threadIdx.x;
    const int wv_  = tid >> 5;
    const int lane = tid & 31;
    const int lm   = lane & 15;
    const int lh   = lane >> 4;

    const int rb = blockIdx.x & 63;              // 64 row-blocks per (b,h)
    const int bh = blockIdx.x >> 6;              // 0..95
    const int h  = bh % NH;
    const int bI = bh / NH;
    const int r0 = rb * 16;

    // ---- Phase 1: scores S = (Q K^T) * 1/sqrt(D); wave owns 128 columns ----
    {
        v8f acc[8] = {};
        const unsigned short* qrow =
            qb + ((size_t)bh * LQ + r0 + lm) * DH + lh * 8;
        for (int d0 = 0; d0 < DH; d0 += 32) {
            BFrag A;
            A.h[0] = *(const v8u16*)(qrow + d0);
            A.h[1] = *(const v8u16*)(qrow + d0 + 16);
#pragma unroll
            for (int t = 0; t < 8; ++t) {
                const int j = wv_ * 128 + t * 16 + lm;      // key index (B's N)
                const unsigned short* kp =
                    kb + ((size_t)bh * LQ + j) * DH + d0 + lh * 8;
                BFrag Bf;
                Bf.h[0] = *(const v8u16*)kp;
                Bf.h[1] = *(const v8u16*)(kp + 16);
                acc[t] = wmma_bf16(A, Bf, acc[t]);
            }
        }
#pragma unroll
        for (int t = 0; t < 8; ++t)
#pragma unroll
            for (int r = 0; r < 8; ++r)
                S[(lh * 8 + r) * LQ + wv_ * 128 + t * 16 + lm] = acc[t][r] * 0.125f;
    }
    __syncthreads();

    // ---- Phase 2: per-row softmax / sparsemax (one wave per 2 rows) --------
    const bool sp = sparse_mask[h] != 0;
    for (int rr = 0; rr < 2; ++rr) {
        const int row = wv_ * 2 + rr;
        float* srow = &S[row * LQ];
        float z[32];
        float mx = -3.4e38f;
#pragma unroll
        for (int i = 0; i < 32; ++i) {
            z[i] = srow[lane + i * 32];
            mx = fmaxf(mx, z[i]);
        }
#pragma unroll
        for (int o = 16; o > 0; o >>= 1) mx = fmaxf(mx, __shfl_xor(mx, o, 32));

        if (sp) {
            // sparsemax: solve sum(max(z - tau, 0)) == 1 by bisection,
            // tau in [max-1, max] (monotone decreasing)
            float lo = mx - 1.0f, hi = mx;
            for (int it = 0; it < 30; ++it) {
                float tau = 0.5f * (lo + hi);
                float s = 0.0f;
#pragma unroll
                for (int i = 0; i < 32; ++i) s += fmaxf(z[i] - tau, 0.0f);
#pragma unroll
                for (int o = 16; o > 0; o >>= 1) s += __shfl_xor(s, o, 32);
                if (s > 1.0f) lo = tau; else hi = tau;
            }
            const float tau = 0.5f * (lo + hi);
            float p[32];
            float s = 0.0f;
#pragma unroll
            for (int i = 0; i < 32; ++i) { p[i] = fmaxf(z[i] - tau, 0.0f); s += p[i]; }
#pragma unroll
            for (int o = 16; o > 0; o >>= 1) s += __shfl_xor(s, o, 32);
            const float inv = (s > 0.0f) ? 1.0f / s : 0.0f;   // s ~= 1; renormalize residual
#pragma unroll
            for (int i = 0; i < 32; ++i) srow[lane + i * 32] = p[i] * inv;
        } else {
            float s = 0.0f;
#pragma unroll
            for (int i = 0; i < 32; ++i) { z[i] = __expf(z[i] - mx); s += z[i]; }
#pragma unroll
            for (int o = 16; o > 0; o >>= 1) s += __shfl_xor(s, o, 32);
            const float inv = 1.0f / s;
#pragma unroll
            for (int i = 0; i < 32; ++i) srow[lane + i * 32] = z[i] * inv;
        }
    }
    __syncthreads();

    // ---- Phase 3: probs fp32 -> bf16 in LDS --------------------------------
#pragma unroll 4
    for (int i = 0; i < 64; ++i) {
        const int idx = tid + i * 256;
        Sh[idx] = f2bf(S[idx]);
    }
    __syncthreads();

    // ---- Phase 4: context = P @ V; each wave sums a 128-wide K slice -------
    {
        v8f acc[4] = {};
        const int kbase = wv_ * 128;
#pragma unroll
        for (int ks = 0; ks < 4; ++ks) {
            const int k0 = kbase + ks * 32;
            BFrag A;
            const unsigned short* ap = &Sh[lm * LQ + k0 + lh * 8];
            A.h[0] = *(const v8u16*)ap;          // ds_load_b128
            A.h[1] = *(const v8u16*)(ap + 16);
#pragma unroll
            for (int j = 0; j < 4; ++j) {
                const unsigned short* vp =
                    vtb + ((size_t)bh * DH + j * 16 + lm) * LQ + k0 + lh * 8;
                BFrag Bf;
                Bf.h[0] = *(const v8u16*)vp;
                Bf.h[1] = *(const v8u16*)(vp + 16);
                acc[j] = wmma_bf16(A, Bf, acc[j]);
            }
        }
        // store wave partials into S (safe: S fp32 no longer read)
        float* Cp = S;                           // viewed as [8][16][64]
#pragma unroll
        for (int j = 0; j < 4; ++j)
#pragma unroll
            for (int r = 0; r < 8; ++r)
                Cp[(wv_ * 16 + lh * 8 + r) * 64 + j * 16 + lm] = acc[j][r];
    }
    __syncthreads();

    // ---- Phase 5: reduce 8 wave partials, write out [B, L, H*D] fp32 -------
#pragma unroll
    for (int i = 0; i < 4; ++i) {
        const int e = tid * 4 + i;               // 0..1023 (16 rows x 64 feats)
        float s = 0.0f;
#pragma unroll
        for (int ww = 0; ww < 8; ++ww) s += S[ww * 1024 + e];
        const int m  = e >> 6;
        const int dd = e & 63;
        out[((size_t)(bI * LQ + r0 + m)) * HID + h * DH + dd] = s;
    }
}

// ---------------------------------------------------------------------------
// Host launcher
// ---------------------------------------------------------------------------
extern "C" void kernel_launch(void* const* d_in, const int* in_sizes, int n_in,
                              void* d_out, int out_size, void* d_ws, size_t ws_size,
                              hipStream_t stream) {
    (void)in_sizes; (void)n_in; (void)out_size; (void)ws_size;

    const float* x    = (const float*)d_in[0];
    const float* Wq   = (const float*)d_in[1];
    const float* bq   = (const float*)d_in[2];
    const float* Wk   = (const float*)d_in[3];
    const float* bk   = (const float*)d_in[4];
    const float* Wv   = (const float*)d_in[5];
    const float* bv   = (const float*)d_in[6];
    const unsigned char* mask = (const unsigned char*)d_in[7];

    char* ws = (char*)d_ws;
    size_t off = 0;
    auto alloc = [&](size_t bytes) { char* p = ws + off; off += (bytes + 255) & ~(size_t)255; return p; };

    unsigned short* xb  = (unsigned short*)alloc((size_t)NTOK * HID * 2);
    unsigned short* wqb = (unsigned short*)alloc((size_t)HID * HID * 2);
    unsigned short* wkb = (unsigned short*)alloc((size_t)HID * HID * 2);
    unsigned short* wvb = (unsigned short*)alloc((size_t)HID * HID * 2);
    unsigned short* qw  = (unsigned short*)alloc((size_t)NTOK * HID * 2);
    unsigned short* kw  = (unsigned short*)alloc((size_t)NTOK * HID * 2);
    unsigned short* vtw = (unsigned short*)alloc((size_t)NTOK * HID * 2);

    const int nx = NTOK * HID;         // 6291456
    const int nw = HID * HID;          // 589824
    cvt_f32_bf16<<<(nx / 8 + 255) / 256, 256, 0, stream>>>(x,  xb,  nx);
    cvt_f32_bf16<<<(nw / 8 + 255) / 256, 256, 0, stream>>>(Wq, wqb, nw);
    cvt_f32_bf16<<<(nw / 8 + 255) / 256, 256, 0, stream>>>(Wk, wkb, nw);
    cvt_f32_bf16<<<(nw / 8 + 255) / 256, 256, 0, stream>>>(Wv, wvb, nw);

    // 6144 wave-tiles (512 M-tiles x 12 N-tiles), 8 waves/block, 3 projections
    qkv_gemm<<<dim3(768, 3), 256, 0, stream>>>(xb, wqb, wkb, wvb,
                                               bq, bk, bv, qw, kw, vtw);

    // one block per (b, h, 16-row query block): 8*12*64 = 6144 blocks
    attn_fused<<<6144, 256, 0, stream>>>(qw, kw, vtw, mask, (float*)d_out);
}
